// CausalSelfAttentionLayer_14164802142352
// MI455X (gfx1250) — matrix operand order repaired and checked by reading, not compiled, not run on hardware
//
#include <hip/hip_runtime.h>

// ---------------------------------------------------------------------------
// CausalSelfAttentionLayer for MI455X (gfx1250, wave32, WMMA).
// H=4096, NQ=32, NKV=8, D=128, B=4, S=1024, T=4096, QKV_N=6144.
// Compute-bound layer (~0.41 TFLOP vs ~20us of HBM traffic at 23.3 TB/s),
// so all GEMMs + attention matmuls use v_wmma_f32_16x16x32_f16.
// ---------------------------------------------------------------------------

#define H_DIM   4096
#define NQH     32
#define NKVH    8
#define HD      128
#define BATCH   4
#define SEQ     1024
#define TTOK    (BATCH * SEQ)           // 4096
#define QKV_N   ((NQH + 2 * NKVH) * HD) // 6144
#define LN_EPS  1e-5f

typedef _Float16 h16;
typedef __attribute__((ext_vector_type(16))) _Float16 v16h;
typedef __attribute__((ext_vector_type(8)))  _Float16 v8h;
typedef __attribute__((ext_vector_type(8)))  float    v8f;

__device__ __forceinline__ v8f wmma_f16(v16h a, v16h b, v8f c) {
  // (neg_a, A, neg_b, B, c_mod, C, reuse_a, reuse_b)
  return __builtin_amdgcn_wmma_f32_16x16x32_f16(false, a, false, b, (short)0, c,
                                                false, false);
}

__device__ __forceinline__ v8f zero8() {
  v8f z;
#pragma unroll
  for (int i = 0; i < 8; ++i) z[i] = 0.0f;
  return z;
}

// Load a 16-half fragment as two contiguous 8-half (16B) chunks, `gap` halves apart.
__device__ __forceinline__ v16h ld_frag(const h16* p, int gap) {
  v8h lo = *(const v8h*)p;
  v8h hi = *(const v8h*)(p + gap);
  v16h r;
#pragma unroll
  for (int i = 0; i < 8; ++i) { r[i] = lo[i]; r[i + 8] = hi[i]; }
  return r;
}

// ---------------------------------------------------------------------------
// Kernel 1: LayerNorm (fp32 in) -> f16 normalized activations.
// ---------------------------------------------------------------------------
__global__ __launch_bounds__(256) void ln_kernel(const float* __restrict__ x,
                                                 const float* __restrict__ g,
                                                 const float* __restrict__ b,
                                                 h16* __restrict__ xn) {
  __shared__ float red[256];
  const int row = blockIdx.x, tid = threadIdx.x;
  const float* xr = x + (size_t)row * H_DIM;

  float s = 0.f;
  for (int i = tid; i < H_DIM; i += 256) s += xr[i];
  red[tid] = s;
  __syncthreads();
  for (int st = 128; st > 0; st >>= 1) {
    if (tid < st) red[tid] += red[tid + st];
    __syncthreads();
  }
  const float mu = red[0] * (1.0f / H_DIM);
  __syncthreads();

  float s2 = 0.f;
  for (int i = tid; i < H_DIM; i += 256) {
    float d = xr[i] - mu;
    s2 += d * d;
  }
  red[tid] = s2;
  __syncthreads();
  for (int st = 128; st > 0; st >>= 1) {
    if (tid < st) red[tid] += red[tid + st];
    __syncthreads();
  }
  const float rstd = rsqrtf(red[0] * (1.0f / H_DIM) + LN_EPS);

  h16* outr = xn + (size_t)row * H_DIM;
  for (int i = tid; i < H_DIM; i += 256)
    outr[i] = (h16)((xr[i] - mu) * rstd * g[i] + b[i]);
}

// ---------------------------------------------------------------------------
// Kernel 2: fp32 -> f16 conversion (weights).
// ---------------------------------------------------------------------------
__global__ void cvt_f16(const float* __restrict__ in, h16* __restrict__ out, int n) {
  int i = blockIdx.x * blockDim.x + threadIdx.x;
  if (i < n) out[i] = (h16)in[i];
}

// ---------------------------------------------------------------------------
// Kernel 3: f16 GEMM  C[M,N] = A[M,K] @ B[K,N] (+bias).
// Block tile 128x64, 128 threads (4 waves); each wave owns a 32x64 strip:
// 2 A-frags x 4 B-frags -> 8 WMMAs per K-step (B-frags reused across both
// M-tiles). A staged to LDS as-is, B staged transposed so every fragment load
// is two contiguous ds_load_b128s giving the same contiguous k order (16*hf + s) in A and B.
// Next K-step tiles are prefetched (global_prefetch_b8) during the WMMA burst.
// ---------------------------------------------------------------------------
template <typename OUT_T, bool BIAS>
__global__ __launch_bounds__(128) void gemm_f16_kernel(
    const h16* __restrict__ A, const h16* __restrict__ B,
    const float* __restrict__ bias, OUT_T* __restrict__ C,
    int M, int N, int K) {
  __shared__ h16 sA[128 * 32];  // [row][k]
  __shared__ h16 sBT[64 * 32];  // [n][k]  (transposed on write)

  const int tid = threadIdx.x, lane = tid & 31, wv = tid >> 5;
  const int m0 = blockIdx.y * 128, n0 = blockIdx.x * 64;
  const int lrow = lane & 15;
  const int hf = (lane < 16) ? 0 : 1;

  v8f acc[2][4];
#pragma unroll
  for (int mt = 0; mt < 2; ++mt)
#pragma unroll
    for (int nt = 0; nt < 4; ++nt) acc[mt][nt] = zero8();

  for (int k0 = 0; k0 < K; k0 += 32) {
    __syncthreads();
    // A tile: 128 rows x 32 k, straight copy (16B vectors).
#pragma unroll
    for (int it = 0; it < 4; ++it) {
      int idx = (tid + it * 128) * 8;  // half offset in tile
      int r = idx >> 5, c = idx & 31;
      *(v8h*)&sA[idx] = *(const v8h*)&A[(size_t)(m0 + r) * K + k0 + c];
    }
    // B tile: 32 k x 64 n, read coalesced, write transposed [n][k].
#pragma unroll
    for (int it = 0; it < 2; ++it) {
      int idx = (tid + it * 128) * 8;
      int r = idx >> 6, c = idx & 63;
      v8h bv = *(const v8h*)&B[(size_t)(k0 + r) * N + n0 + c];
#pragma unroll
      for (int i = 0; i < 8; ++i) sBT[(c + i) * 32 + r] = bv[i];
    }
    __syncthreads();

    // Prefetch next K-step tiles into cache while WMMAs run.
    if (k0 + 32 < K) {
      {
        int idx = tid * 8;
        int r = idx >> 5, c = idx & 31;
        __builtin_prefetch(&A[(size_t)(m0 + r) * K + k0 + 32 + c], 0, 3);
      }
      {
        int idx = tid * 8;
        int r = idx >> 6, c = idx & 63;
        __builtin_prefetch(&B[(size_t)(k0 + 32 + r) * N + n0 + c], 0, 3);
      }
    }

    v16h af[2];
#pragma unroll
    for (int mt = 0; mt < 2; ++mt) {
      int arow = wv * 32 + mt * 16 + lrow;
      af[mt] = ld_frag(&sA[arow * 32 + hf * 16], 8);  // A: 16 contiguous K per lane
    }
#pragma unroll
    for (int nt = 0; nt < 4; ++nt) {
      int n = nt * 16 + lrow;
      v16h bf = ld_frag(&sBT[n * 32 + hf * 16], 8);  // B: 16 contiguous K per lane
      acc[0][nt] = wmma_f16(af[0], bf, acc[0][nt]);
      acc[1][nt] = wmma_f16(af[1], bf, acc[1][nt]);
    }
  }

  // C layout: lane holds col n0+lrow, rows r + 8*hf (r = VGPR index).
#pragma unroll
  for (int mt = 0; mt < 2; ++mt) {
    const int rbase = m0 + wv * 32 + mt * 16 + hf * 8;
#pragma unroll
    for (int nt = 0; nt < 4; ++nt) {
      int col = n0 + nt * 16 + lrow;
      float bia = BIAS ? bias[col] : 0.0f;
#pragma unroll
      for (int r = 0; r < 8; ++r) {
        C[(size_t)(rbase + r) * N + col] = (OUT_T)(acc[mt][nt][r] + bia);
      }
    }
  }
}

// ---------------------------------------------------------------------------
// Kernel 4: RoPE + layout prep.
//   q -> [T][NQ*D] f16 (rotated)
//   k -> [B][NKV][S][D] f16 (rotated)   v -> [B][NKV][S][D] f16
// ---------------------------------------------------------------------------
__global__ __launch_bounds__(256) void rope_prep_kernel(
    const h16* __restrict__ qkv, h16* __restrict__ qf,
    h16* __restrict__ kf, h16* __restrict__ vf) {
  const int t = blockIdx.x;
  const int b = t / SEQ, pos = t % SEQ;
  const h16* src = qkv + (size_t)t * QKV_N;

  for (int i = threadIdx.x; i < 3584; i += 256) {
    if (i < 2048) {  // Q rotation pairs: NQ heads x 64 pairs
      int h = i >> 6, j = i & 63;
      float x1 = (float)src[h * HD + j];
      float x2 = (float)src[h * HD + 64 + j];
      float ang = (float)pos * __powf(10000.0f, -(float)j / 64.0f);
      float sn, cs;
      __sincosf(ang, &sn, &cs);
      size_t base = (size_t)t * (NQH * HD) + h * HD;
      qf[base + j]      = (h16)(x1 * cs - x2 * sn);
      qf[base + 64 + j] = (h16)(x1 * sn + x2 * cs);
    } else if (i < 2560) {  // K rotation pairs: NKV heads x 64 pairs
      int ii = i - 2048, h = ii >> 6, j = ii & 63;
      float x1 = (float)src[NQH * HD + h * HD + j];
      float x2 = (float)src[NQH * HD + h * HD + 64 + j];
      float ang = (float)pos * __powf(10000.0f, -(float)j / 64.0f);
      float sn, cs;
      __sincosf(ang, &sn, &cs);
      size_t base = ((size_t)(b * NKVH + h) * SEQ + pos) * HD;
      kf[base + j]      = (h16)(x1 * cs - x2 * sn);
      kf[base + 64 + j] = (h16)(x1 * sn + x2 * cs);
    } else {  // V copy: NKV heads x 128
      int ii = i - 2560, h = ii >> 7, j = ii & 127;
      size_t base = ((size_t)(b * NKVH + h) * SEQ + pos) * HD;
      vf[base + j] = src[(NQH + NKVH) * HD + h * HD + j];
    }
  }
}

// ---------------------------------------------------------------------------
// Kernel 5: Flash attention (causal, GQA rep=4).
// Block = (batch, head, 64 q-rows); 4 waves of 16 rows; KV streamed in blocks
// of 32 keys staged through LDS. Uniform KV trip count per block (masking
// handles causality) so __syncthreads() inside the loop is legal.
// ---------------------------------------------------------------------------
__global__ __launch_bounds__(128) void attn_kernel(
    const h16* __restrict__ q,  // [T][NQ*D]
    const h16* __restrict__ k,  // [B][NKV][S][D]
    const h16* __restrict__ v,  // [B][NKV][S][D]
    h16* __restrict__ o) {      // [T][NQ*D]
  __shared__ h16 sK[32 * 128];   // [key][d]     (QK^T B-frags: contiguous d)
  __shared__ h16 sVT[128 * 32];  // [d][key]     (PV  B-frags: contiguous key)
  __shared__ h16 sP[4][16 * 32]; // per-wave P (C-layout -> A-frag relayout)

  const int tid = threadIdx.x, lane = tid & 31, wv = tid >> 5;
  const int qblk = blockIdx.x, head = blockIdx.y, batch = blockIdx.z;
  const int hkv = head >> 2;  // NQ/NKV = 4
  const int qbase = qblk * 64;
  const int lrow = lane & 15;
  const int hf = (lane < 16) ? 0 : 1;
  const float scale = 0.088388347648318447f;  // 1/sqrt(128)

  // Preload this wave's 16x128 Q tile as 4 A-fragments (global b128 loads).
  const h16* qp =
      q + (size_t)(batch * SEQ + qbase + wv * 16 + lrow) * (NQH * HD) + head * HD;
  v16h qfr[4];
#pragma unroll
  for (int kk = 0; kk < 4; ++kk) qfr[kk] = ld_frag(qp + kk * 32 + hf * 16, 8);

  float mrun[8], lrun[8];
  v8f oacc[8];
#pragma unroll
  for (int r = 0; r < 8; ++r) { mrun[r] = -1e30f; lrun[r] = 0.0f; }
#pragma unroll
  for (int dt = 0; dt < 8; ++dt) oacc[dt] = zero8();

  const h16* kg = k + (size_t)(batch * NKVH + hkv) * SEQ * HD;
  const h16* vg = v + (size_t)(batch * NKVH + hkv) * SEQ * HD;

  const int nkvblk = (qbase + 64) / 32;  // uniform across the block
  for (int kvb = 0; kvb < nkvblk; ++kvb) {
    const int kv0 = kvb * 32;
    __syncthreads();
    // Stage K (straight copy) and V (transposed) : 4096 halves each.
#pragma unroll
    for (int it = 0; it < 4; ++it) {
      int idx = (tid + it * 128) * 8;
      *(v8h*)&sK[idx] = *(const v8h*)&kg[(size_t)kv0 * HD + idx];
      v8h vv = *(const v8h*)&vg[(size_t)kv0 * HD + idx];
      int key = idx >> 7, d0 = idx & 127;
#pragma unroll
      for (int i = 0; i < 8; ++i) sVT[(d0 + i) * 32 + key] = vv[i];
    }
    __syncthreads();

    // S = Q @ K^T for two 16-key n-tiles.
    v8f s0 = zero8(), s1 = zero8();
#pragma unroll
    for (int kk = 0; kk < 4; ++kk) {
      v16h b0 = ld_frag(&sK[lrow * 128 + kk * 32 + hf * 16], 8);
      v16h b1 = ld_frag(&sK[(16 + lrow) * 128 + kk * 32 + hf * 16], 8);
      s0 = wmma_f16(qfr[kk], b0, s0);
      s1 = wmma_f16(qfr[kk], b1, s1);
    }

    // Online softmax. Row of element (lane, VGPR r) = r + 8*hf; col = lrow.
    const int kc0 = kv0 + lrow, kc1 = kv0 + 16 + lrow;
    float fsc[8];
#pragma unroll
    for (int r = 0; r < 8; ++r) {
      const int qpos = qbase + wv * 16 + r + hf * 8;
      float x0 = (kc0 <= qpos) ? s0[r] * scale : -1e30f;
      float x1 = (kc1 <= qpos) ? s1[r] * scale : -1e30f;
      float mx = fmaxf(x0, x1);
#pragma unroll
      for (int off = 1; off < 16; off <<= 1)
        mx = fmaxf(mx, __shfl_xor(mx, off, 32));
      float mnew = fmaxf(mrun[r], mx);
      float f = __expf(mrun[r] - mnew);
      float e0 = __expf(x0 - mnew);
      float e1 = __expf(x1 - mnew);
      float rs = e0 + e1;
#pragma unroll
      for (int off = 1; off < 16; off <<= 1) rs += __shfl_xor(rs, off, 32);
      mrun[r] = mnew;
      lrun[r] = lrun[r] * f + rs;
      fsc[r] = f;
      // Scatter P (f16) into per-wave LDS in plain [row][key] order.
      const int row = r + hf * 8;
      sP[wv][row * 32 + lrow]      = (h16)e0;
      sP[wv][row * 32 + 16 + lrow] = (h16)e1;
    }

    // Rescale running O (row factors are per-lane -> pure VALU).
#pragma unroll
    for (int dt = 0; dt < 8; ++dt)
#pragma unroll
      for (int r = 0; r < 8; ++r) oacc[dt][r] *= fsc[r];

    // Wave-local store->load through LDS: DS ops are in-order per wave; make
    // the dependency explicit for the scheduler.
    asm volatile("s_wait_dscnt 0" ::: "memory");

    v16h pf = ld_frag(&sP[wv][lrow * 32 + hf * 16], 8);  // P as A-fragment
#pragma unroll
    for (int dt = 0; dt < 8; ++dt) {
      v16h vfab = ld_frag(&sVT[(dt * 16 + lrow) * 32 + hf * 16], 8);
      oacc[dt] = wmma_f16(pf, vfab, oacc[dt]);
    }
  }

  // Normalize and write O (f16, [T][NQ*D]).
  h16* op = o + (size_t)(batch * SEQ + qbase + wv * 16) * (NQH * HD) + head * HD;
#pragma unroll
  for (int r = 0; r < 8; ++r) {
    const float inv = 1.0f / lrun[r];
    const int row = r + hf * 8;
#pragma unroll
    for (int dt = 0; dt < 8; ++dt)
      op[(size_t)row * (NQH * HD) + dt * 16 + lrow] = (h16)(oacc[dt][r] * inv);
  }
}

// ---------------------------------------------------------------------------
// Host-side orchestration (graph-capture safe: kernel launches only).
// Workspace layout (bytes), total ~240 MB:
//   xn_f16     : 32 MB   w_qkv_f16 : 48 MB   qkv_f16 : 48 MB
//   q_f16      : 32 MB   k_f16     :  8 MB   v_f16   :  8 MB
//   o_f16      : 32 MB   w_proj_f16: 32 MB
// ---------------------------------------------------------------------------
extern "C" void kernel_launch(void* const* d_in, const int* in_sizes, int n_in,
                              void* d_out, int out_size, void* d_ws, size_t ws_size,
                              hipStream_t stream) {
  const float* hidden = (const float*)d_in[0];
  // d_in[1] cu_seqlens, d_in[2] max_seqlen: fixed-shape problem (B=4, S=1024).
  const float* ln_g   = (const float*)d_in[3];
  const float* ln_b   = (const float*)d_in[4];
  const float* w_qkv  = (const float*)d_in[5];
  const float* b_qkv  = (const float*)d_in[6];
  const float* w_proj = (const float*)d_in[7];
  const float* b_proj = (const float*)d_in[8];
  float* out = (float*)d_out;

  char* ws = (char*)d_ws;
  size_t off = 0;
  h16* xn     = (h16*)(ws + off); off += (size_t)TTOK * H_DIM * 2;     // 32 MB
  h16* wqkv_h = (h16*)(ws + off); off += (size_t)H_DIM * QKV_N * 2;    // 48 MB
  h16* qkv_h  = (h16*)(ws + off); off += (size_t)TTOK * QKV_N * 2;     // 48 MB
  h16* q_h    = (h16*)(ws + off); off += (size_t)TTOK * NQH * HD * 2;  // 32 MB
  h16* k_h    = (h16*)(ws + off); off += (size_t)BATCH * NKVH * SEQ * HD * 2;
  h16* v_h    = (h16*)(ws + off); off += (size_t)BATCH * NKVH * SEQ * HD * 2;
  h16* o_h    = (h16*)(ws + off); off += (size_t)TTOK * NQH * HD * 2;  // 32 MB
  h16* wproj_h= (h16*)(ws + off); off += (size_t)H_DIM * H_DIM * 2;    // 32 MB
  (void)ws_size; (void)in_sizes; (void)n_in; (void)out_size;

  // 1) LayerNorm -> f16
  ln_kernel<<<TTOK, 256, 0, stream>>>(hidden, ln_g, ln_b, xn);

  // 2) Weight down-conversion
  {
    int n1 = H_DIM * QKV_N;
    cvt_f16<<<(n1 + 255) / 256, 256, 0, stream>>>(w_qkv, wqkv_h, n1);
    int n2 = H_DIM * H_DIM;
    cvt_f16<<<(n2 + 255) / 256, 256, 0, stream>>>(w_proj, wproj_h, n2);
  }

  // 3) QKV projection: [T,H] @ [H,6144] + bias -> f16
  gemm_f16_kernel<h16, true><<<dim3(QKV_N / 64, TTOK / 128), 128, 0, stream>>>(
      xn, wqkv_h, b_qkv, qkv_h, TTOK, QKV_N, H_DIM);

  // 4) RoPE + q/k/v layout prep
  rope_prep_kernel<<<TTOK, 256, 0, stream>>>(qkv_h, q_h, k_h, v_h);

  // 5) Flash attention
  attn_kernel<<<dim3(SEQ / 64, NQH, BATCH), 128, 0, stream>>>(q_h, k_h, v_h, o_h);

  // 6) Output projection: [T,4096] @ [4096,4096] + bias -> fp32 out
  gemm_f16_kernel<float, true><<<dim3(H_DIM / 64, TTOK / 128), 128, 0, stream>>>(
      o_h, wproj_h, b_proj, out, TTOK, H_DIM, H_DIM);
}
